// MHSA_45938970198397
// MI455X (gfx1250) — compile-verified, matrix operand order checked by reading, not run on hardware
//
#include <hip/hip_runtime.h>
#include <math.h>

#define B_ 2
#define L_ 2048
#define C_ 1024
#define H_ 16
#define D_ 64

typedef __attribute__((ext_vector_type(16))) _Float16 v16h;
typedef __attribute__((ext_vector_type(8)))  _Float16 v8h;
typedef __attribute__((ext_vector_type(4)))  _Float16 v4h;
typedef __attribute__((ext_vector_type(8)))  float    v8f;
typedef __attribute__((ext_vector_type(4)))  int      v4i_t;

__device__ __forceinline__ v16h cat16(v8h lo, v8h hi) {
  return __builtin_shufflevector(lo, hi, 0,1,2,3,4,5,6,7,8,9,10,11,12,13,14,15);
}

__device__ __forceinline__ v8f wmma_f16(v16h a, v16h b, v8f c) {
  // D = A(16x32 f16) * B(32x16 f16) + C(16x16 f32)
  return __builtin_amdgcn_wmma_f32_16x16x32_f16(false, a, false, b, (short)0, c,
                                                false, false);
}

// DPP16 move: CTRL 0x121/0x122/0x124/0x128 = row_ror:1/2/4/8 (row = 16 lanes)
template <int CTRL>
__device__ __forceinline__ float dpp_mv(float x) {
  return __int_as_float(__builtin_amdgcn_update_dpp(
      0, __float_as_int(x), CTRL, 0xf, 0xf, true));
}

// max across the 16-lane row (VALU-only rotate reduction, no LDS)
__device__ __forceinline__ float rowmax16(float x) {
  x = fmaxf(x, dpp_mv<0x121>(x));
  x = fmaxf(x, dpp_mv<0x122>(x));
  x = fmaxf(x, dpp_mv<0x124>(x));
  x = fmaxf(x, dpp_mv<0x128>(x));
  return x;
}

// -------- gfx1250 async global->LDS copy (guarded; falls back to sync) -----
#if __has_builtin(__builtin_amdgcn_global_load_async_to_lds_b128)
#define HAVE_ASYNC_LDS 1
#else
#define HAVE_ASYNC_LDS 0
#endif

__device__ __forceinline__ void cp128_g2l(const _Float16* g, _Float16* l) {
#if HAVE_ASYNC_LDS
  __builtin_amdgcn_global_load_async_to_lds_b128((v4i_t*)g, (v4i_t*)l, 0, 0);
#else
  *(v8h*)l = *(const v8h*)g;
#endif
}

__device__ __forceinline__ void wait_async_lds() {
#if HAVE_ASYNC_LDS && __has_builtin(__builtin_amdgcn_s_wait_asynccnt)
  __builtin_amdgcn_s_wait_asynccnt(0);
#endif
}

// ---------------------------------------------------------------------------
// LayerNorm: x[4096,1024] f32 -> xn f16.  One block (256 thr) per row.
// ---------------------------------------------------------------------------
__global__ __launch_bounds__(256) void ln_kernel(
    const float* __restrict__ x, const float* __restrict__ gamma,
    const float* __restrict__ beta, _Float16* __restrict__ xn) {
  const int row = blockIdx.x;
  const int t = threadIdx.x;
  const float* xr = x + (size_t)row * C_;
  float4 v = *(const float4*)(xr + t * 4);
  float s  = v.x + v.y + v.z + v.w;
  float s2 = v.x * v.x + v.y * v.y + v.z * v.z + v.w * v.w;
  #pragma unroll
  for (int m = 1; m < 32; m <<= 1) {
    s  += __shfl_xor(s,  m, 32);
    s2 += __shfl_xor(s2, m, 32);
  }
  __shared__ float red0[8], red1[8];
  const int lane = t & 31, wid = t >> 5;
  if (lane == 0) { red0[wid] = s; red1[wid] = s2; }
  __syncthreads();
  if (t == 0) {
    float a = 0.f, b = 0.f;
    #pragma unroll
    for (int i = 0; i < 8; i++) { a += red0[i]; b += red1[i]; }
    red0[0] = a; red1[0] = b;
  }
  __syncthreads();
  const float mean = red0[0] * (1.0f / C_);
  const float var  = red1[0] * (1.0f / C_) - mean * mean;
  const float rstd = rsqrtf(var + 1e-5f);
  const int c0 = t * 4;
  float xv[4] = {v.x, v.y, v.z, v.w};
  v4h o;
  #pragma unroll
  for (int i = 0; i < 4; i++)
    o[i] = (_Float16)(((xv[i] - mean) * rstd) * gamma[c0 + i] + beta[c0 + i]);
  *(v4h*)(xn + (size_t)row * C_ + c0) = o;
}

// ---------------------------------------------------------------------------
// f32 -> f16 weight conversion (vectorized by 4)
// ---------------------------------------------------------------------------
__global__ __launch_bounds__(256) void cvt_kernel(const float* __restrict__ s,
                                                  _Float16* __restrict__ d, int n4) {
  int i = blockIdx.x * blockDim.x + threadIdx.x;
  const int stride = gridDim.x * blockDim.x;
  for (; i < n4; i += stride) {
    float4 f = ((const float4*)s)[i];
    v4h o = {(_Float16)f.x, (_Float16)f.y, (_Float16)f.z, (_Float16)f.w};
    ((v4h*)d)[i] = o;
  }
}

// ---------------------------------------------------------------------------
// GEMM: out[M,N] = A[M,K]*W[K,N] + bias.   A,W f16 row-major.
// Block = 256 thr (8 waves). Block tile 256(M) x 64(N); wave tile 32 x 64.
// K stepped by 64: 16 WMMAs per barrier.  W tile (64k x 64n) double-buffered
// TRANSPOSED in LDS ([n][k]); next tile's global loads issued before the
// WMMA burst.  OUT_MODE 0: f16 out into [B,H,L,D].  OUT_MODE 1: f32 [M,N].
// ---------------------------------------------------------------------------
template <int OUT_MODE>
__global__ __launch_bounds__(256) void gemm_kernel(
    const _Float16* __restrict__ A, const _Float16* __restrict__ W,
    const float* __restrict__ bias, _Float16* __restrict__ out16,
    float* __restrict__ out32, int M, int N, int K) {
  __shared__ _Float16 wt[2][64 * 64];  // double-buffered W tile [n][k]
  const int t = threadIdx.x, lane = t & 31;
  const int wid = t >> 5;
  const int lr = lane & 15;
  const int hi16 = lane >> 4;
  const int off = hi16 * 8;          // K-pattern {0..7,16..23} / {8..15,24..31}
  const int n0 = blockIdx.x * 64;
  const int m0 = blockIdx.y * 256 + wid * 32;
  const int sk = t >> 3;             // 0..31 (k row within half-tile)
  const int sn = (t & 7) * 8;        // n chunk
  v8f acc[2][4] = {};
  const _Float16* a0p = A + (size_t)(m0 + lr) * K + off;
  const _Float16* a1p = A + (size_t)(m0 + 16 + lr) * K + off;
  const _Float16* wp  = W + (size_t)sk * N + n0 + sn;
  // prologue: stage k0=0 tile (64 k-rows) into buffer 0
  {
    v8h w0 = *(const v8h*)wp;
    v8h w1 = *(const v8h*)(wp + (size_t)32 * N);
    #pragma unroll
    for (int j = 0; j < 8; j++) {
      wt[0][(sn + j) * 64 + sk]      = w0[j];
      wt[0][(sn + j) * 64 + 32 + sk] = w1[j];
    }
  }
  __syncthreads();
  for (int k0 = 0; k0 < K; k0 += 64) {
    const int p = (k0 >> 6) & 1;
    const bool nxt = (k0 + 64 < K);
    v8h wn0, wn1;
    if (nxt) {                                     // early global loads
      wn0 = *(const v8h*)(wp + (size_t)(k0 + 64) * N);
      wn1 = *(const v8h*)(wp + (size_t)(k0 + 96) * N);
    }
    if (k0 + 128 < K) {
      __builtin_prefetch(a0p + k0 + 128, 0, 3);
      __builtin_prefetch(a1p + k0 + 128, 0, 3);
    }
    v16h a0k0 = cat16(*(const v8h*)(a0p + k0),      *(const v8h*)(a0p + k0 + 16));
    v16h a0k1 = cat16(*(const v8h*)(a0p + k0 + 32), *(const v8h*)(a0p + k0 + 48));
    v16h a1k0 = cat16(*(const v8h*)(a1p + k0),      *(const v8h*)(a1p + k0 + 16));
    v16h a1k1 = cat16(*(const v8h*)(a1p + k0 + 32), *(const v8h*)(a1p + k0 + 48));
    #pragma unroll
    for (int tt = 0; tt < 4; tt++) {
      const _Float16* bp = &wt[p][(tt * 16 + lr) * 64 + off];
      v16h blo = cat16(*(const v8h*)bp,        *(const v8h*)(bp + 16));
      v16h bhi = cat16(*(const v8h*)(bp + 32), *(const v8h*)(bp + 48));
      acc[0][tt] = wmma_f16(a0k0, blo, acc[0][tt]);
      acc[0][tt] = wmma_f16(a0k1, bhi, acc[0][tt]);
      acc[1][tt] = wmma_f16(a1k0, blo, acc[1][tt]);
      acc[1][tt] = wmma_f16(a1k1, bhi, acc[1][tt]);
    }
    if (nxt) {
      #pragma unroll
      for (int j = 0; j < 8; j++) {
        wt[1 - p][(sn + j) * 64 + sk]      = wn0[j];
        wt[1 - p][(sn + j) * 64 + 32 + sk] = wn1[j];
      }
    }
    __syncthreads();
  }
  #pragma unroll
  for (int r = 0; r < 2; r++) {
    #pragma unroll
    for (int tt = 0; tt < 4; tt++) {
      const int n = n0 + tt * 16 + lr;
      const float bv = bias[n];
      #pragma unroll
      for (int i = 0; i < 8; i++) {
        const int m = m0 + r * 16 + i + hi16 * 8;  // C/D: lanes16-31 = rows +8
        const float val = acc[r][tt][i] + bv;
        if (OUT_MODE == 0) {
          const int b = m >> 11, l = m & (L_ - 1);
          const int h = n >> 6, d = n & (D_ - 1);
          out16[((((size_t)b * H_ + h) * L_ + l) << 6) + d] = (_Float16)val;
        } else {
          out32[(size_t)m * N + n] = val;
        }
      }
    }
  }
}

// ---------------------------------------------------------------------------
// Flash attention.  One block = 8 waves, ALL on the same (b,h); wave w owns
// 16 query rows; block covers 128 q rows.  Keys chunked by 64.
// K / V^T chunks double-buffered in LDS; K staged with async b128 global->LDS
// (next chunk issued before computing the current one) -> ONE block barrier
// per chunk.  P transpose slab is wave-private (wave_barrier only).
// 18 WMMAs / wave / chunk: 8 scores + 8 P*V + 2 row-sum (P * ones).
// ---------------------------------------------------------------------------
__global__ __launch_bounds__(256) void attn_kernel(
    const _Float16* __restrict__ q, const _Float16* __restrict__ k,
    const _Float16* __restrict__ v, _Float16* __restrict__ attn) {
  __shared__ _Float16 kbuf[2][64 * 64];  // K chunk [key][d]      (2 x 8 KB)
  __shared__ _Float16 vbufT[2][64 * 64]; // V chunk [d][key]      (2 x 8 KB)
  __shared__ _Float16 pbuf[8][16 * 64];  // per-wave P slabs      (16 KB)
  const int t = threadIdx.x, lane = t & 31, wid = t >> 5;
  const int qblk = blockIdx.x & (L_ / 128 - 1);
  const int h    = (blockIdx.x >> 4) & (H_ - 1);
  const int b    = blockIdx.x >> 8;
  const int lr = lane & 15;
  const int off = (lane >> 4) * 8;
  const size_t head = ((size_t)b * H_ + h) * L_ * D_;
  const _Float16* qg = q + head;
  const _Float16* kg = k + head;
  const _Float16* vg = v + head;
  const int l0 = qblk * 128 + wid * 16;
  // staging: thread t covers key = t%64, d-chunks sd and sd+32
  const int skey = t & 63;
  const int sd   = (t >> 6) * 8;     // 0,8,16,24

  // Q as A-operand, pre-scaled by 1/sqrt(D)=0.125 (exact in f16)
  const _Float16* qr = qg + (size_t)(l0 + lr) * D_ + off;
  v16h aq0 = cat16(*(const v8h*)qr,        *(const v8h*)(qr + 16));
  v16h aq1 = cat16(*(const v8h*)(qr + 32), *(const v8h*)(qr + 48));
  v16h ones;
  #pragma unroll
  for (int i = 0; i < 16; i++) {
    aq0[i] *= (_Float16)0.125f;
    aq1[i] *= (_Float16)0.125f;
    ones[i] = (_Float16)1.0f;
  }

  float m_i[8];
  v8f o[4] = {};
  v8f lacc = {};                 // running row sums of P, via WMMA with ones-B
  #pragma unroll
  for (int i = 0; i < 8; i++) m_i[i] = -1e30f;
  _Float16* pslab = pbuf[wid];

  // ---- prologue: stage chunk 0 into buffer 0 ----
  {
    const size_t base = (size_t)skey * D_;
    cp128_g2l(kg + base + sd,      &kbuf[0][skey * 64 + sd]);
    cp128_g2l(kg + base + sd + 32, &kbuf[0][skey * 64 + sd + 32]);
    v8h va = *(const v8h*)(vg + base + sd);
    v8h vc = *(const v8h*)(vg + base + sd + 32);
    #pragma unroll
    for (int j = 0; j < 8; j++) {
      vbufT[0][(sd + j) * 64 + skey]      = va[j];
      vbufT[0][(sd + 32 + j) * 64 + skey] = vc[j];
    }
  }
  wait_async_lds();
  __syncthreads();

  for (int kc = 0; kc < L_; kc += 64) {
    const int p = (kc >> 6) & 1;
    const bool nxt = (kc + 64 < L_);
    v8h va, vc;
    if (nxt) {
      // issue next chunk's loads before computing the current one
      const size_t base = (size_t)(kc + 64 + skey) * D_;
      cp128_g2l(kg + base + sd,      &kbuf[1 - p][skey * 64 + sd]);
      cp128_g2l(kg + base + sd + 32, &kbuf[1 - p][skey * 64 + sd + 32]);
      va = *(const v8h*)(vg + base + sd);
      vc = *(const v8h*)(vg + base + sd + 32);
    }
    if (kc + 128 < L_) {
      __builtin_prefetch(kg + (size_t)(kc + 128 + skey) * D_ + sd, 0, 3);
      __builtin_prefetch(vg + (size_t)(kc + 128 + skey) * D_ + sd, 0, 3);
    }
    // ---- scores: 4 key-tiles x (2 WMMAs over D=64) ----
    v8f s[4];
    #pragma unroll
    for (int tt = 0; tt < 4; tt++) {
      const _Float16* kr = &kbuf[p][(tt * 16 + lr) * 64 + off];
      v16h blo = cat16(*(const v8h*)kr,        *(const v8h*)(kr + 16));
      v16h bhi = cat16(*(const v8h*)(kr + 32), *(const v8h*)(kr + 48));
      v8f z = {};
      z = wmma_f16(aq0, blo, z);
      s[tt] = wmma_f16(aq1, bhi, z);
    }
    // ---- online softmax over 64 new cols (C/D layout; 16-lane rows) ----
    float pv[4][8];
    #pragma unroll
    for (int i = 0; i < 8; i++) {
      const float mx01 = fmaxf(s[0][i], s[1][i]);
      const float mx23 = fmaxf(s[2][i], s[3][i]);
      const float mx = rowmax16(fmaxf(mx01, mx23));
      const float mn = fmaxf(m_i[i], mx);
      const float corr = __expf(m_i[i] - mn);
      m_i[i] = mn;
      #pragma unroll
      for (int tt = 0; tt < 4; tt++) pv[tt][i] = __expf(s[tt][i] - mn);
      lacc[i] *= corr;
      #pragma unroll
      for (int tt = 0; tt < 4; tt++) o[tt][i] *= corr;
    }
    // ---- transpose P (D-layout -> A-layout) via wave-private LDS slab ----
    #pragma unroll
    for (int i = 0; i < 8; i++) {
      const int r = i + off;
      pslab[r * 64 + lr]      = (_Float16)pv[0][i];
      pslab[r * 64 + 16 + lr] = (_Float16)pv[1][i];
      pslab[r * 64 + 32 + lr] = (_Float16)pv[2][i];
      pslab[r * 64 + 48 + lr] = (_Float16)pv[3][i];
    }
    __builtin_amdgcn_wave_barrier();   // wave-local write->read ordering
    const _Float16* pr = &pslab[lr * 64 + off];
    v16h pa0 = cat16(*(const v8h*)pr,        *(const v8h*)(pr + 16));
    v16h pa1 = cat16(*(const v8h*)(pr + 32), *(const v8h*)(pr + 48));
    // ---- row sums: lacc += P * ones (both key halves) ----
    lacc = wmma_f16(pa0, ones, lacc);
    lacc = wmma_f16(pa1, ones, lacc);
    // ---- O += P * V : 4 d-tiles x 2 key-halves, all ds_load_b128 ----
    #pragma unroll
    for (int tt = 0; tt < 4; tt++) {
      const _Float16* vp = &vbufT[p][(tt * 16 + lr) * 64];
      v16h bvlo = cat16(*(const v8h*)(vp + off),      *(const v8h*)(vp + 16 + off));
      v16h bvhi = cat16(*(const v8h*)(vp + 32 + off), *(const v8h*)(vp + 48 + off));
      o[tt] = wmma_f16(pa0, bvlo, o[tt]);
      o[tt] = wmma_f16(pa1, bvhi, o[tt]);
    }
    // ---- scatter next V chunk into the other buffer ----
    if (nxt) {
      #pragma unroll
      for (int j = 0; j < 8; j++) {
        vbufT[1 - p][(sd + j) * 64 + skey]      = va[j];
        vbufT[1 - p][(sd + 32 + j) * 64 + skey] = vc[j];
      }
    }
    wait_async_lds();
    __syncthreads();   // single block barrier per chunk
  }
  // ---- epilogue: normalize, store f16 into [B*L, C] ----
  #pragma unroll
  for (int i = 0; i < 8; i++) {
    const float inv = 1.0f / lacc[i];
    const int r = i + off;
    const size_t row = (size_t)b * L_ + l0 + r;
    #pragma unroll
    for (int tt = 0; tt < 4; tt++)
      attn[row * C_ + h * D_ + tt * 16 + lr] = (_Float16)(o[tt][i] * inv);
  }
}

// ---------------------------------------------------------------------------
extern "C" void kernel_launch(void* const* d_in, const int* in_sizes, int n_in,
                              void* d_out, int out_size, void* d_ws, size_t ws_size,
                              hipStream_t stream) {
  const float* x    = (const float*)d_in[0];
  const float* ln_g = (const float*)d_in[1];
  const float* ln_b = (const float*)d_in[2];
  const float* wq   = (const float*)d_in[3];
  const float* bq   = (const float*)d_in[4];
  const float* wk   = (const float*)d_in[5];
  const float* bk   = (const float*)d_in[6];
  const float* wv   = (const float*)d_in[7];
  const float* bv   = (const float*)d_in[8];
  const float* wo   = (const float*)d_in[9];
  const float* bo   = (const float*)d_in[10];

  char* ws = (char*)d_ws;
  _Float16* xn   = (_Float16*)ws;                       // 4M elems, 8 MB
  _Float16* wq16 = (_Float16*)(ws + (8u << 20));        // 1M elems each
  _Float16* wk16 = wq16 + (1u << 20);
  _Float16* wv16 = wk16 + (1u << 20);
  _Float16* wo16 = wv16 + (1u << 20);
  _Float16* qb   = (_Float16*)(ws + (16u << 20));       // 4M elems each
  _Float16* kb   = qb + (4u << 20);
  _Float16* vb   = kb + (4u << 20);
  _Float16* ab   = vb + (4u << 20);

  const int M = B_ * L_;   // 4096
  const int N = C_;        // 1024
  const int K = C_;        // 1024

  ln_kernel<<<M, 256, 0, stream>>>(x, ln_g, ln_b, xn);
  cvt_kernel<<<512, 256, 0, stream>>>(wq, wq16, (C_ * C_) / 4);
  cvt_kernel<<<512, 256, 0, stream>>>(wk, wk16, (C_ * C_) / 4);
  cvt_kernel<<<512, 256, 0, stream>>>(wv, wv16, (C_ * C_) / 4);
  cvt_kernel<<<512, 256, 0, stream>>>(wo, wo16, (C_ * C_) / 4);

  dim3 g(N / 64, M / 256);
  gemm_kernel<0><<<g, 256, 0, stream>>>(xn, wq16, bq, qb, nullptr, M, N, K);
  gemm_kernel<0><<<g, 256, 0, stream>>>(xn, wk16, bk, kb, nullptr, M, N, K);
  gemm_kernel<0><<<g, 256, 0, stream>>>(xn, wv16, bv, vb, nullptr, M, N, K);

  // blocks: B * H * (L/128) = 2*16*16 = 512, each 8 waves on one (b,h)
  attn_kernel<<<B_ * H_ * (L_ / 128), 256, 0, stream>>>(qb, kb, vb, ab);

  gemm_kernel<1><<<g, 256, 0, stream>>>(ab, wo16, bo, nullptr, (float*)d_out, M, N, K);
}
